// DeformableAttentionWithSpconv_14499809592084
// MI455X (gfx1250) — matrix operand order repaired and checked by reading, not compiled, not run on hardware
//
#include <hip/hip_runtime.h>
#include <hip/hip_bf16.h>

// ---------- problem constants (match reference) ----------
#define GX 128
#define GY 128
#define GZ 16
#define BB 2
#define VV 100000
#define NN 8192
#define KK 8
#define CC 64
#define MTOT (BB * VV)          // 200000 voxel rows
#define QROWS (BB * NN)         // 16384 query rows
#define STAT_BLOCKS 256

typedef __attribute__((ext_vector_type(16))) _Float16 v16h;
typedef __attribute__((ext_vector_type(8)))  float    v8f;
typedef __attribute__((ext_vector_type(4)))  unsigned int v4u;

union Frag { v16h h; v4u q[2]; };

__device__ inline v16h zero16() {
    v16h r;
#pragma unroll
    for (int i = 0; i < 16; ++i) r[i] = (_Float16)0.0f;
    return r;
}
__device__ inline v8f zero8() {
    v8f r;
#pragma unroll
    for (int i = 0; i < 8; ++i) r[i] = 0.0f;
    return r;
}

// ---------- 1. grid hash ----------
__global__ void init_grid_kernel(int* __restrict__ grid, int n) {
    int t = blockIdx.x * blockDim.x + threadIdx.x;
    if (t < n) grid[t] = -1;
}

__global__ void scatter_grid_kernel(const int* __restrict__ coords, int* __restrict__ grid) {
    int r = blockIdx.x * blockDim.x + threadIdx.x;
    if (r >= MTOT) return;
    int b = r / VV;
    int x = coords[r * 3 + 0], y = coords[r * 3 + 1], z = coords[r * 3 + 2];
    grid[((b * GX + x) * GY + y) * GZ + z] = r;
}

// ---------- 2. weight pre-swizzle into WMMA B-fragment layout ----------
// fragment f = ((k*2 + kc)*4 + nt); within: lane holds col N = nt*16 + (lane&15),
// 16 halves = K = kc*32 + ((lane&16)?8:0) + (h&7) + ((h&8)?16:0)  -> contiguous 32B per lane.
__global__ void prep_conv_w_kernel(const float* __restrict__ W, _Float16* __restrict__ Wp) {
    int t = blockIdx.x * blockDim.x + threadIdx.x;          // 27*4096 threads
    if (t >= 27 * 4096) return;
    int h = t & 15, lane = (t >> 4) & 31, frag = t >> 9;
    int nt = frag & 3, kc = (frag >> 2) & 1, k = frag >> 3;
    int Kd = kc * 32 + ((lane & 16) ? 8 : 0) + (h & 7) + ((h & 8) ? 16 : 0);
    int Nd = nt * 16 + (lane & 15);
    Wp[t] = (_Float16)W[(k * 64 + Kd) * 64 + Nd];           // W[k][cin][cout]
}

__global__ void prep_proj_w_kernel(const float* __restrict__ PW, _Float16* __restrict__ Pp) {
    int t = blockIdx.x * blockDim.x + threadIdx.x;          // 4096 threads
    if (t >= 4096) return;
    int h = t & 15, lane = (t >> 4) & 31, frag = t >> 9;
    int nt = frag & 3, kc = frag >> 2;
    int Kd = kc * 32 + ((lane & 16) ? 8 : 0) + (h & 7) + ((h & 8) ? 16 : 0);          // cin
    int Nd = nt * 16 + (lane & 15);                         // cout
    Pp[t] = (_Float16)PW[Nd * 64 + Kd];                     // B = proj_W^T
}

// ---------- 3. f32 -> f16 feature convert ----------
__global__ void cvt_f16_kernel(const float* __restrict__ in, _Float16* __restrict__ out, size_t n) {
    size_t t = (size_t)blockIdx.x * blockDim.x + threadIdx.x;
    if (t < n) out[t] = (_Float16)in[t];
}

// ---------- 4. submanifold conv via WMMA ----------
// one wave = 16 rows x 64 cols; 27 taps x 2 K-chunks x 4 N-tiles = 216 wmma
__global__ __launch_bounds__(256)
void subm_conv_wmma_kernel(const _Float16* __restrict__ featH,
                           const _Float16* __restrict__ Wp,
                           const int* __restrict__ grid,
                           const int* __restrict__ coords,
                           float* __restrict__ out) {
    int tile = blockIdx.x * 8 + (threadIdx.x >> 5);
    int lane = threadIdx.x & 31;
    int rowBase = tile * 16;
    if (rowBase >= MTOT) return;                 // wave-uniform: EXEC stays all-ones
    int myRow = rowBase + (lane & 15);
    bool rv = myRow < MTOT;
    int rr = rv ? myRow : (MTOT - 1);
    int cx = coords[rr * 3 + 0], cy = coords[rr * 3 + 1], cz = coords[rr * 3 + 2];
    int b = rr / VV;
    int sub = (lane & 16) ? 8 : 0;

    v8f acc[4];
#pragma unroll
    for (int n = 0; n < 4; ++n) acc[n] = zero8();

    for (int k = 0; k < 27; ++k) {
        int dx = k / 9 - 1, dy = (k / 3) % 3 - 1, dz = k % 3 - 1;
        int nx = cx + dx, ny = cy + dy, nz = cz + dz;
        bool ok = rv && nx >= 0 && nx < GX && ny >= 0 && ny < GY && nz >= 0 && nz < GZ;
        int nidx = -1;
        if (ok) nidx = grid[((b * GX + nx) * GY + ny) * GZ + nz];

#pragma unroll
        for (int c = 0; c < 2; ++c) {
            v16h a;
            if (nidx >= 0) {
                const _Float16* p = featH + (size_t)nidx * 64 + c * 32 + sub;
                Frag u;
                u.q[0] = *(const v4u*)p;          // K = c0+sub .. +7
                u.q[1] = *(const v4u*)(p + 16);   // K = c0+16+sub .. +7
                a = u.h;
            } else {
                a = zero16();
            }
            const _Float16* wb = Wp + (size_t)((k * 2 + c) * 4) * 512 + lane * 16;
#pragma unroll
            for (int n = 0; n < 4; ++n) {
                Frag ub;
                const _Float16* q = wb + n * 512;
                ub.q[0] = *(const v4u*)q;
                ub.q[1] = *(const v4u*)(q + 8);
                acc[n] = __builtin_amdgcn_wmma_f32_16x16x32_f16(
                    false, a, false, ub.h, (short)0, acc[n], false, false);
            }
        }
    }

#pragma unroll
    for (int n = 0; n < 4; ++n) {
#pragma unroll
        for (int r = 0; r < 8; ++r) {
            int orow = rowBase + ((lane & 16) ? r + 8 : r);
            if (orow < MTOT) out[(size_t)orow * 64 + n * 16 + (lane & 15)] = acc[n][r];
        }
    }
}

// ---------- 5. deterministic BN stats ----------
// per-block fixed-order partials -> fixed-order final reduce (no float atomics)
__global__ __launch_bounds__(256)
void bn_partials_kernel(const float* __restrict__ x, float* __restrict__ partials) {
    __shared__ float s[256], s2[256];
    int c = threadIdx.x & 63;
    int j = threadIdx.x >> 6;                    // 4 substreams per channel
    const int rowsPerBlock = (MTOT + STAT_BLOCKS - 1) / STAT_BLOCKS;
    int r0 = blockIdx.x * rowsPerBlock;
    int r1 = min(MTOT, r0 + rowsPerBlock);
    float a = 0.0f, b = 0.0f;
    for (int r = r0 + j; r < r1; r += 4) {
        float v = x[(size_t)r * 64 + c];
        a += v; b += v * v;
    }
    s[threadIdx.x] = a; s2[threadIdx.x] = b;
    __syncthreads();
    if (j == 0) {
        float sa = s[c] + s[64 + c] + s[128 + c] + s[192 + c];
        float sb = s2[c] + s2[64 + c] + s2[128 + c] + s2[192 + c];
        partials[(size_t)blockIdx.x * 128 + c] = sa;
        partials[(size_t)blockIdx.x * 128 + 64 + c] = sb;
    }
}

__global__ void bn_reduce_kernel(const float* __restrict__ partials, float* __restrict__ sums) {
    int i = threadIdx.x;                         // 128 threads
    if (i >= 128) return;
    float s = 0.0f;
    for (int blk = 0; blk < STAT_BLOCKS; ++blk) s += partials[(size_t)blk * 128 + i];
    sums[i] = s;
}

// ---------- 6. BN apply + ReLU (+ f16 requant, optional f32 copy) ----------
__global__ void bn_apply_kernel(const float* __restrict__ x, const float* __restrict__ sums,
                                const float* __restrict__ gamma, const float* __restrict__ beta,
                                _Float16* __restrict__ outH, float* __restrict__ outF) {
    size_t t = (size_t)blockIdx.x * blockDim.x + threadIdx.x;
    if (t >= (size_t)MTOT * 64) return;
    int c = (int)(t & 63);
    const float invM = 1.0f / (float)MTOT;
    float mu = sums[c] * invM;
    float var = sums[64 + c] * invM - mu * mu;
    float y = (x[t] - mu) * rsqrtf(var + 1e-5f) * gamma[c] + beta[c];
    y = fmaxf(y, 0.0f);
    outH[t] = (_Float16)y;
    if (outF) outF[t] = y;
}

// ---------- 7. keypoint quantize + index output ----------
__global__ void quantize_kernel(const float* __restrict__ kp, const int* __restrict__ grid,
                                int* __restrict__ idxBuf, float* __restrict__ outIdx) {
    int t = blockIdx.x * blockDim.x + threadIdx.x;   // B*N*K = 131072
    if (t >= BB * NN * KK) return;
    float x = kp[t * 3 + 0], y = kp[t * 3 + 1], z = kp[t * 3 + 2];
    int qx = min(max((int)(x * 2.0f), 0), GX - 1);   // 1/VSIZE = 2
    int qy = min(max((int)(y * 2.0f), 0), GY - 1);
    int qz = min(max((int)(z * 2.0f), 0), GZ - 1);
    int b = t / (NN * KK);
    int idx = grid[((b * GX + qx) * GY + qy) * GZ + qz];
    idxBuf[t] = idx < 0 ? 0 : idx;
    float* o = outIdx + (size_t)t * 4;
    o[0] = (float)b; o[1] = (float)qx; o[2] = (float)qy; o[3] = (float)qz;
}

// ---------- 8. gather + mean + query add (f16 for proj WMMA) ----------
__global__ void fuse_gather_kernel(const float* __restrict__ featF, const int* __restrict__ idxBuf,
                                   const float* __restrict__ query, _Float16* __restrict__ fusedIn) {
    size_t t = (size_t)blockIdx.x * blockDim.x + threadIdx.x;   // QROWS*64
    if (t >= (size_t)QROWS * 64) return;
    int c = (int)(t & 63);
    int bn = (int)(t >> 6);
    const int* ib = idxBuf + (size_t)bn * 8;
    float s = 0.0f;
#pragma unroll
    for (int k = 0; k < 8; ++k) s += featF[(size_t)ib[k] * 64 + c];
    fusedIn[t] = (_Float16)(s * 0.125f + query[t]);
}

// ---------- 9. projection GEMM via WMMA (+ bias) ----------
__global__ __launch_bounds__(256)
void proj_wmma_kernel(const _Float16* __restrict__ in, const _Float16* __restrict__ Pp,
                      const float* __restrict__ bias, float* __restrict__ out) {
    int tile = blockIdx.x * 8 + (threadIdx.x >> 5);
    int lane = threadIdx.x & 31;
    int rowBase = tile * 16;
    if (rowBase >= QROWS) return;                // QROWS % 16 == 0, no ragged edge
    int myRow = rowBase + (lane & 15);
    int sub = (lane & 16) ? 8 : 0;

    v8f acc[4];
#pragma unroll
    for (int n = 0; n < 4; ++n) acc[n] = zero8();

#pragma unroll
    for (int c = 0; c < 2; ++c) {
        const _Float16* p = in + (size_t)myRow * 64 + c * 32 + sub;
        Frag ua;
        ua.q[0] = *(const v4u*)p;
        ua.q[1] = *(const v4u*)(p + 16);
        const _Float16* wb = Pp + (size_t)(c * 4) * 512 + lane * 16;
#pragma unroll
        for (int n = 0; n < 4; ++n) {
            Frag ub;
            const _Float16* q = wb + n * 512;
            ub.q[0] = *(const v4u*)q;
            ub.q[1] = *(const v4u*)(q + 8);
            acc[n] = __builtin_amdgcn_wmma_f32_16x16x32_f16(
                false, ua.h, false, ub.h, (short)0, acc[n], false, false);
        }
    }

#pragma unroll
    for (int n = 0; n < 4; ++n) {
#pragma unroll
        for (int r = 0; r < 8; ++r) {
            int orow = rowBase + ((lane & 16) ? r + 8 : r);
            int ocol = n * 16 + (lane & 15);
            out[(size_t)orow * 64 + ocol] = acc[n][r] + bias[ocol];
        }
    }
}

// =====================================================================
extern "C" void kernel_launch(void* const* d_in, const int* in_sizes, int n_in,
                              void* d_out, int out_size, void* d_ws, size_t ws_size,
                              hipStream_t stream) {
    const float* keypoints = (const float*)d_in[0];
    const float* query     = (const float*)d_in[1];
    const float* voxfeat   = (const float*)d_in[2];
    const int*   voxcoord  = (const int*)d_in[3];
    const float* W1        = (const float*)d_in[4];
    const float* g1        = (const float*)d_in[5];
    const float* b1        = (const float*)d_in[6];
    const float* W2        = (const float*)d_in[7];
    const float* g2        = (const float*)d_in[8];
    const float* b2        = (const float*)d_in[9];
    const float* projW     = (const float*)d_in[10];
    const float* projB     = (const float*)d_in[11];

    float* out      = (float*)d_out;
    float* outFused = out;                                   // B*N*C floats
    float* outIdx   = out + (size_t)BB * NN * CC;            // B*N*K*4 coord values

    // workspace carve-up (all offsets 256B-aligned)
    char* ws = (char*)d_ws;
    size_t off = 0;
    int*       grid     = (int*)(ws + off);       off += (size_t)BB * GX * GY * GZ * 4;       // 2 MB
    _Float16*  featH0   = (_Float16*)(ws + off);  off += (size_t)MTOT * 64 * 2;               // 25.6 MB
    _Float16*  featH1   = (_Float16*)(ws + off);  off += (size_t)MTOT * 64 * 2;               // 25.6 MB
    float*     convOut  = (float*)(ws + off);     off += (size_t)MTOT * 64 * 4;               // 51.2 MB
    float*     featF    = (float*)(ws + off);     off += (size_t)MTOT * 64 * 4;               // 51.2 MB
    _Float16*  W1p      = (_Float16*)(ws + off);  off += (size_t)27 * 4096 * 2;               // 221 KB
    _Float16*  W2p      = (_Float16*)(ws + off);  off += (size_t)27 * 4096 * 2;
    _Float16*  Pp       = (_Float16*)(ws + off);  off += (size_t)4096 * 2;
    float*     partials = (float*)(ws + off);     off += (size_t)STAT_BLOCKS * 128 * 4;       // 128 KB
    float*     sums     = (float*)(ws + off);     off += 512;
    int*       idxBuf   = (int*)(ws + off);       off += (size_t)BB * NN * KK * 4;            // 512 KB
    _Float16*  fusedIn  = (_Float16*)(ws + off);  off += (size_t)QROWS * 64 * 2;              // 2 MB

    const int gridCells = BB * GX * GY * GZ;
    const size_t featElems = (size_t)MTOT * 64;
    const int convBlocks = ( (MTOT + 15) / 16 + 7 ) / 8;     // 8 waves (16-row tiles) / block
    const int projBlocks = ( (QROWS + 15) / 16 + 7 ) / 8;

    // 1. build grid hash
    init_grid_kernel<<<(gridCells + 255) / 256, 256, 0, stream>>>(grid, gridCells);
    scatter_grid_kernel<<<(MTOT + 255) / 256, 256, 0, stream>>>(voxcoord, grid);

    // 2. pre-swizzle weights into WMMA B-fragment layout
    prep_conv_w_kernel<<<(27 * 4096 + 255) / 256, 256, 0, stream>>>(W1, W1p);
    prep_conv_w_kernel<<<(27 * 4096 + 255) / 256, 256, 0, stream>>>(W2, W2p);
    prep_proj_w_kernel<<<(4096 + 255) / 256, 256, 0, stream>>>(projW, Pp);

    // 3. features f32 -> f16
    cvt_f16_kernel<<<(int)((featElems + 255) / 256), 256, 0, stream>>>(voxfeat, featH0, featElems);

    // 4. layer 1: conv -> BN stats -> BN+ReLU (f16)
    subm_conv_wmma_kernel<<<convBlocks, 256, 0, stream>>>(featH0, W1p, grid, voxcoord, convOut);
    bn_partials_kernel<<<STAT_BLOCKS, 256, 0, stream>>>(convOut, partials);
    bn_reduce_kernel<<<1, 128, 0, stream>>>(partials, sums);
    bn_apply_kernel<<<(int)((featElems + 255) / 256), 256, 0, stream>>>(
        convOut, sums, g1, b1, featH1, nullptr);

    // 5. layer 2: conv -> BN stats -> BN+ReLU (f16 scratch + f32 for sampling)
    subm_conv_wmma_kernel<<<convBlocks, 256, 0, stream>>>(featH1, W2p, grid, voxcoord, convOut);
    bn_partials_kernel<<<STAT_BLOCKS, 256, 0, stream>>>(convOut, partials);
    bn_reduce_kernel<<<1, 128, 0, stream>>>(partials, sums);
    bn_apply_kernel<<<(int)((featElems + 255) / 256), 256, 0, stream>>>(
        convOut, sums, g2, b2, featH0, featF);

    // 6. keypoint quantize (+ coordinate output), gather+mean+query, projection
    quantize_kernel<<<(BB * NN * KK + 255) / 256, 256, 0, stream>>>(keypoints, grid, idxBuf, outIdx);
    fuse_gather_kernel<<<(int)(((size_t)QROWS * 64 + 255) / 256), 256, 0, stream>>>(
        featF, idxBuf, query, fusedIn);
    proj_wmma_kernel<<<projBlocks, 256, 0, stream>>>(fusedIn, Pp, projB, outFused);

    (void)in_sizes; (void)n_in; (void)out_size; (void)ws_size;
}
